// EpisodicMemory_82867099009522
// MI455X (gfx1250) — matrix operand order repaired and checked by reading, not compiled, not run on hardware
//
#include <hip/hip_runtime.h>

#define BSZ     4
#define NSEQ    1024
#define NBH     8
#define DDIM    64
#define MMEM    2048
#define KRET    16
#define CCAND   64
#define NEGV    -1.0e9f
#define EPSV    1e-8f
#define SMAXV   4.0f
#define BUDGETV 512.0f

typedef __attribute__((ext_vector_type(16))) __bf16 v16bf;
typedef __attribute__((ext_vector_type(8)))  float  v8f;
typedef __attribute__((ext_vector_type(4)))  int    v4i;

union B16V { v16bf v; uint4 q[2]; unsigned short u[16]; };

#if defined(__has_builtin)
#if __has_builtin(__builtin_amdgcn_global_load_async_to_lds_b128) && \
    __has_builtin(__builtin_amdgcn_s_wait_asynccnt)
#define HAVE_ASYNC_LDS 1
#endif
#endif

#ifdef HAVE_ASYNC_LDS
#define AS1q __attribute__((address_space(1)))
#define AS3q __attribute__((address_space(3)))
__device__ __forceinline__ void async_g2l_b128(const void* g, void* l) {
  __builtin_amdgcn_global_load_async_to_lds_b128((AS1q v4i*)(g), (AS3q v4i*)(l), 0, 0);
}
#endif

__device__ __forceinline__ unsigned short f2bf(float x) {
  unsigned int t = __float_as_uint(x);
  unsigned int r = t + 0x7FFFu + ((t >> 16) & 1u);   // round-to-nearest-even
  return (unsigned short)(r >> 16);
}

__device__ __forceinline__ v8f vzero() {
  v8f z = {0.f, 0.f, 0.f, 0.f, 0.f, 0.f, 0.f, 0.f};
  return z;
}

__device__ __forceinline__ v8f wmma_bf16(v16bf a, v16bf b, v8f c) {
  return __builtin_amdgcn_wmma_f32_16x16x32_bf16(false, a, false, b, (short)0, c, false, false);
}

// ---------------------------------------------------------------------------
// Kernel 1: convert em_K -> bf16 [sb][m][d]; em_V -> bf16 transposed [sb][d][m]
// ---------------------------------------------------------------------------
__global__ void k_convert(const float* __restrict__ emK, const float* __restrict__ emV,
                          unsigned short* __restrict__ emKh, unsigned short* __restrict__ emVhT) {
  const int total = BSZ * NBH * MMEM * DDIM;
  for (int i = blockIdx.x * blockDim.x + threadIdx.x; i < total; i += gridDim.x * blockDim.x) {
    emKh[i] = f2bf(emK[i]);
    const int d  = i % DDIM;
    const int m  = (i / DDIM) % MMEM;
    const int sb = i / (DDIM * MMEM);
    emVhT[((size_t)sb * DDIM + d) * MMEM + m] = f2bf(emV[i]);
  }
}

// ---------------------------------------------------------------------------
// Kernel 2: fused read (top-k masked attention) + novelty. 1 wave = 16 q rows.
// ---------------------------------------------------------------------------
__global__ void __launch_bounds__(32) k_attn(
    const float* __restrict__ q, const float* __restrict__ qn,
    const float* __restrict__ surprise, const float* __restrict__ wnov,
    const float* __restrict__ emS,
    const unsigned short* __restrict__ emKh, const unsigned short* __restrict__ emVhT,
    float* __restrict__ outp, float* __restrict__ nov) {
  const int lane = threadIdx.x;
  const int ntile = blockIdx.x & 63;
  const int sb = blockIdx.x >> 6;
  const int s = sb >> 3, b = sb & 7;
  const int n0 = ntile * 16;
  const int col = lane & 15;
  const int hi = lane >> 4;
  const int rowoff = hi * 8;

  __shared__ float sc[16][17];
  __shared__ float thr[16], mxv[16], ssum[16];
  __shared__ unsigned short etile[16][34];
  __shared__ __align__(16) float qlds[16 * DDIM];
  __shared__ __align__(16) float nlds[16 * DDIM];

  // ---- stage q / q_nov 16x64 f32 tiles into LDS (async path on gfx1250) ----
  {
    const float* qbase = q  + (((size_t)s * NSEQ + n0) * NBH + b) * DDIM;
    const float* nbase = qn + (((size_t)s * NSEQ + n0) * NBH + b) * DDIM;
    const int rr = lane >> 4;          // 0..1: row parity this lane copies
    const int cc = (lane & 15) * 4;    // float offset (16B granules)
#ifdef HAVE_ASYNC_LDS
#pragma unroll
    for (int i = 0; i < 8; ++i) {
      const int row = i * 2 + rr;
      async_g2l_b128(qbase + (size_t)row * NBH * DDIM + cc, qlds + row * DDIM + cc);
      async_g2l_b128(nbase + (size_t)row * NBH * DDIM + cc, nlds + row * DDIM + cc);
    }
    __builtin_amdgcn_s_wait_asynccnt(0);
#else
#pragma unroll
    for (int i = 0; i < 8; ++i) {
      const int row = i * 2 + rr;
      *(float4*)(qlds + row * DDIM + cc) = *(const float4*)(qbase + (size_t)row * NBH * DDIM + cc);
      *(float4*)(nlds + row * DDIM + cc) = *(const float4*)(nbase + (size_t)row * NBH * DDIM + cc);
    }
#endif
  }
  __syncthreads();

  // A operands for q and q_nov (16x32 bf16 per k-step), built once from LDS.
  B16V aq[2], an[2];
  {
#pragma unroll
    for (int ks = 0; ks < 2; ++ks) {
      const int d0 = ks * 32 + hi * 16;
#pragma unroll
      for (int j = 0; j < 8; ++j) {
        aq[ks].u[j]     = f2bf(qlds[col * DDIM + d0 + j]);
        aq[ks].u[j + 8] = f2bf(qlds[col * DDIM + d0 + 8 + j]);
        an[ks].u[j]     = f2bf(nlds[col * DDIM + d0 + j]);
        an[ks].u[j + 8] = f2bf(nlds[col * DDIM + d0 + 8 + j]);
      }
    }
  }

  float t16[KRET];
#pragma unroll
  for (int i = 0; i < KRET; ++i) t16[i] = NEGV;
  float smax[8];
#pragma unroll
  for (int r = 0; r < 8; ++r) smax[r] = -1.0f;

  const unsigned short* Kb = emKh + (size_t)sb * MMEM * DDIM;
  const unsigned short* Vb = emVhT + (size_t)sb * DDIM * MMEM;
  const float* Sb = emS + (size_t)sb * MMEM;

  // ---- pass 1: scores (top-16) + sim-max ----
  for (int mt = 0; mt < MMEM / 16; ++mt) {
    const int m = mt * 16 + col;
    const unsigned short* krow = Kb + (size_t)m * DDIM;
    __builtin_prefetch(krow + 16 * DDIM, 0, 1);
    B16V bk[2];
#pragma unroll
    for (int ks = 0; ks < 2; ++ks) {
      const unsigned short* p = krow + ks * 32 + hi * 16;
      bk[ks].q[0] = *(const uint4*)(p);
      bk[ks].q[1] = *(const uint4*)(p + 8);
    }
    v8f accS = vzero(), accN = vzero();
    accS = wmma_bf16(aq[0].v, bk[0].v, accS);
    accS = wmma_bf16(aq[1].v, bk[1].v, accS);
    accN = wmma_bf16(an[0].v, bk[0].v, accN);
    accN = wmma_bf16(an[1].v, bk[1].v, accN);
    const bool act = Sb[m] > 0.0f;
#pragma unroll
    for (int r = 0; r < 8; ++r) {
      sc[r + rowoff][col] = act ? accS[r] : NEGV;
      smax[r] = fmaxf(smax[r], act ? accN[r] : -1.0f);
    }
    __syncthreads();
    if (lane < 16) {
#pragma unroll
      for (int j = 0; j < 16; ++j) {
        float v = sc[lane][j];
        if (v > t16[KRET - 1]) {
          t16[KRET - 1] = v;
#pragma unroll
          for (int i = KRET - 1; i > 0; --i) {
            float a = t16[i - 1], c = t16[i];
            t16[i - 1] = fmaxf(a, c);
            t16[i] = fminf(a, c);
          }
        }
      }
    }
    __syncthreads();
  }

  if (lane < 16) { thr[lane] = t16[KRET - 1]; mxv[lane] = t16[0]; }

  // novelty: reduce sim-max across 16-lane group, write [sb][n]
#pragma unroll
  for (int r = 0; r < 8; ++r) {
    float v = smax[r];
#pragma unroll
    for (int k = 1; k < 16; k <<= 1) v = fmaxf(v, __shfl_xor(v, k, 32));
    smax[r] = v;
  }
  if (col == 0) {
#pragma unroll
    for (int r = 0; r < 8; ++r) {
      const int n = n0 + r + rowoff;
      const float ms = fmaxf(smax[r], 0.0f);
      const float w  = wnov[((size_t)s * NSEQ + n) * NBH + b];
      const float sp = surprise[((size_t)s * NSEQ + n) * NBH + b];
      nov[(size_t)sb * NSEQ + n] = w * sp + (1.0f - w) * (1.0f - ms);
    }
  }
  __syncthreads();

  // ---- pass 2: masked exp -> attn (bf16 A) x em_V^T (bf16 B) ----
  v8f accO[4];
#pragma unroll
  for (int dt = 0; dt < 4; ++dt) accO[dt] = vzero();
  float psum[8];
#pragma unroll
  for (int r = 0; r < 8; ++r) psum[r] = 0.0f;

  for (int mp = 0; mp < MMEM / 32; ++mp) {
#pragma unroll
    for (int half = 0; half < 2; ++half) {
      const int mt = 2 * mp + half;
      const int m = mt * 16 + col;
      const unsigned short* krow = Kb + (size_t)m * DDIM;
      B16V bk[2];
#pragma unroll
      for (int ks = 0; ks < 2; ++ks) {
        const unsigned short* p = krow + ks * 32 + hi * 16;
        bk[ks].q[0] = *(const uint4*)(p);
        bk[ks].q[1] = *(const uint4*)(p + 8);
      }
      v8f accS = vzero();
      accS = wmma_bf16(aq[0].v, bk[0].v, accS);
      accS = wmma_bf16(aq[1].v, bk[1].v, accS);
      const bool act = Sb[m] > 0.0f;
#pragma unroll
      for (int r = 0; r < 8; ++r) {
        const int row = r + rowoff;
        const float sv = act ? accS[r] : NEGV;
        const float e = (sv >= thr[row]) ? __expf(sv - mxv[row]) : 0.0f;
        psum[r] += e;
        etile[row][half * 16 + col] = f2bf(e);
      }
    }
    __syncthreads();
    B16V ae;
    {
      const int k0 = hi * 16;
#pragma unroll
      for (int j = 0; j < 8; ++j) {
        ae.u[j]     = etile[col][k0 + j];
        ae.u[j + 8] = etile[col][k0 + 8 + j];
      }
    }
#pragma unroll
    for (int dt = 0; dt < 4; ++dt) {
      const unsigned short* vrow = Vb + (size_t)(dt * 16 + col) * MMEM + mp * 32 + hi * 16;
      B16V bv;
      bv.q[0] = *(const uint4*)(vrow);
      bv.q[1] = *(const uint4*)(vrow + 8);
      accO[dt] = wmma_bf16(ae.v, bv.v, accO[dt]);
    }
    __syncthreads();
  }

#pragma unroll
  for (int r = 0; r < 8; ++r) {
    float v = psum[r];
#pragma unroll
    for (int k = 1; k < 16; k <<= 1) v += __shfl_xor(v, k, 32);
    if (col == 0) ssum[r + rowoff] = v;
  }
  __syncthreads();
#pragma unroll
  for (int dt = 0; dt < 4; ++dt) {
#pragma unroll
    for (int r = 0; r < 8; ++r) {
      const int row = r + rowoff, d = dt * 16 + col;
      outp[(((size_t)s * NSEQ + n0 + row) * NBH + b) * DDIM + d] =
          accO[dt][r] / fmaxf(ssum[row], 1e-20f);
    }
  }
}

// ---------------------------------------------------------------------------
// Kernel 3: top-64 candidates per (s,b); emit bf16 ck (row + T) and cand_V^T
// ---------------------------------------------------------------------------
__global__ void __launch_bounds__(32) k_cand(
    const float* __restrict__ nov, const float* __restrict__ qn, const float* __restrict__ vn,
    unsigned short* __restrict__ ckh, unsigned short* __restrict__ ckhT,
    unsigned short* __restrict__ cvhT, float* __restrict__ candw, float* __restrict__ totS) {
  const int sb = blockIdx.x;
  const int s = sb >> 3, b = sb & 7;
  const int lane = threadIdx.x;
  __shared__ float sval[NSEQ];
  __shared__ int cidx[CCAND];
  __shared__ float cscore[CCAND];
  __shared__ float wsum;

  for (int i = lane; i < NSEQ; i += 32) sval[i] = nov[(size_t)sb * NSEQ + i];
  if (lane == 0) totS[sb] = 0.0f;
  __syncthreads();

  for (int c = 0; c < CCAND; ++c) {
    float best = -1e30f; int bi = 0;
    for (int i = lane; i < NSEQ; i += 32) {
      const float v = sval[i];
      if (v > best) { best = v; bi = i; }
    }
    for (int k = 16; k > 0; k >>= 1) {
      const float ov = __shfl_xor(best, k, 32);
      const int oi = __shfl_xor(bi, k, 32);
      if (ov > best || (ov == best && oi < bi)) { best = ov; bi = oi; }
    }
    if (lane == 0) { cidx[c] = bi; cscore[c] = best; sval[bi] = -1e30f; }
    __syncthreads();
  }
  if (lane == 0) {
    float acc = 0.0f;
    for (int c = 0; c < CCAND; ++c) acc += cscore[c];
    wsum = acc + EPSV;
  }
  __syncthreads();
  for (int c = lane; c < CCAND; c += 32) candw[(size_t)sb * CCAND + c] = cscore[c] / wsum;

  for (int c = 0; c < CCAND; ++c) {
    const int n = cidx[c];
    const float* kr = qn + (((size_t)s * NSEQ + n) * NBH + b) * DDIM;
    const float* vr = vn + (((size_t)s * NSEQ + n) * NBH + b) * DDIM;
    const float x0 = kr[lane], x1 = kr[lane + 32];
    float nrm = x0 * x0 + x1 * x1;
    for (int k = 16; k > 0; k >>= 1) nrm += __shfl_xor(nrm, k, 32);
    const float inv = 1.0f / fmaxf(sqrtf(nrm), EPSV);
    const unsigned short h0 = f2bf(x0 * inv), h1 = f2bf(x1 * inv);
    ckh[((size_t)sb * CCAND + c) * DDIM + lane]       = h0;
    ckh[((size_t)sb * CCAND + c) * DDIM + lane + 32]  = h1;
    ckhT[((size_t)sb * DDIM + lane) * CCAND + c]      = h0;
    ckhT[((size_t)sb * DDIM + lane + 32) * CCAND + c] = h1;
    cvhT[((size_t)sb * DDIM + lane) * CCAND + c]      = f2bf(vr[lane]);
    cvhT[((size_t)sb * DDIM + lane + 32) * CCAND + c] = f2bf(vr[lane + 32]);
  }
}

// ---------------------------------------------------------------------------
// Kernel 4: per-(sb, M-chunk) softmax partials (rowmax, rowsumexp) over slots
// ---------------------------------------------------------------------------
__global__ void __launch_bounds__(32) k_stats1(
    const unsigned short* __restrict__ emKh, const float* __restrict__ emS,
    const unsigned short* __restrict__ ckh,
    const float* __restrict__ tau, const float* __restrict__ wwp,
    float* __restrict__ pmax, float* __restrict__ psum) {
  const int lane = threadIdx.x;
  const int chunk = blockIdx.x & 15;
  const int sb = blockIdx.x >> 4;
  const int col = lane & 15, hi = lane >> 4, rowoff = hi * 8;
  const float tauc = fmaxf(tau[sb], 0.01f);
  const float wws = wwp[sb];
  const unsigned short* Kb = emKh + (size_t)sb * MMEM * DDIM;
  const float* Sb = emS + (size_t)sb * MMEM;

  B16V ak[4][2];
#pragma unroll
  for (int ct = 0; ct < 4; ++ct) {
    const unsigned short* crow = ckh + ((size_t)sb * CCAND + ct * 16 + col) * DDIM;
#pragma unroll
    for (int ks = 0; ks < 2; ++ks) {
      const int d0 = ks * 32 + hi * 16;
#pragma unroll
      for (int j = 0; j < 8; ++j) {
        ak[ct][ks].u[j]     = crow[d0 + j];
        ak[ct][ks].u[j + 8] = crow[d0 + 8 + j];
      }
    }
  }

  __shared__ float cmax[CCAND];
  float pm[4][8];
#pragma unroll
  for (int ct = 0; ct < 4; ++ct)
#pragma unroll
    for (int r = 0; r < 8; ++r) pm[ct][r] = -1e30f;

  for (int t = 0; t < 8; ++t) {
    const int m = (chunk * 8 + t) * 16 + col;
    const unsigned short* krow = Kb + (size_t)m * DDIM;
    B16V bk[2];
#pragma unroll
    for (int ks = 0; ks < 2; ++ks) {
      const unsigned short* p = krow + ks * 32 + hi * 16;
      bk[ks].q[0] = *(const uint4*)(p);
      bk[ks].q[1] = *(const uint4*)(p + 8);
    }
    const float negS = -Sb[m] * wws;
#pragma unroll
    for (int ct = 0; ct < 4; ++ct) {
      v8f acc = vzero();
      acc = wmma_bf16(ak[ct][0].v, bk[0].v, acc);
      acc = wmma_bf16(ak[ct][1].v, bk[1].v, acc);
#pragma unroll
      for (int r = 0; r < 8; ++r)
        pm[ct][r] = fmaxf(pm[ct][r], (acc[r] + negS) / tauc);
    }
  }
#pragma unroll
  for (int ct = 0; ct < 4; ++ct) {
#pragma unroll
    for (int r = 0; r < 8; ++r) {
      float v = pm[ct][r];
#pragma unroll
      for (int k = 1; k < 16; k <<= 1) v = fmaxf(v, __shfl_xor(v, k, 32));
      if (col == 0) cmax[ct * 16 + r + rowoff] = v;
    }
  }
  __syncthreads();

  float ps[4][8];
#pragma unroll
  for (int ct = 0; ct < 4; ++ct)
#pragma unroll
    for (int r = 0; r < 8; ++r) ps[ct][r] = 0.0f;

  for (int t = 0; t < 8; ++t) {
    const int m = (chunk * 8 + t) * 16 + col;
    const unsigned short* krow = Kb + (size_t)m * DDIM;
    B16V bk[2];
#pragma unroll
    for (int ks = 0; ks < 2; ++ks) {
      const unsigned short* p = krow + ks * 32 + hi * 16;
      bk[ks].q[0] = *(const uint4*)(p);
      bk[ks].q[1] = *(const uint4*)(p + 8);
    }
    const float negS = -Sb[m] * wws;
#pragma unroll
    for (int ct = 0; ct < 4; ++ct) {
      v8f acc = vzero();
      acc = wmma_bf16(ak[ct][0].v, bk[0].v, acc);
      acc = wmma_bf16(ak[ct][1].v, bk[1].v, acc);
#pragma unroll
      for (int r = 0; r < 8; ++r) {
        const int c = ct * 16 + r + rowoff;
        ps[ct][r] += __expf((acc[r] + negS) / tauc - cmax[c]);
      }
    }
  }
#pragma unroll
  for (int ct = 0; ct < 4; ++ct) {
#pragma unroll
    for (int r = 0; r < 8; ++r) {
      float v = ps[ct][r];
#pragma unroll
      for (int k = 1; k < 16; k <<= 1) v += __shfl_xor(v, k, 32);
      if (col == 0) {
        const int c = ct * 16 + r + rowoff;
        const int idx = ((sb * 16) + chunk) * CCAND + c;
        psum[idx] = v;
        pmax[idx] = cmax[c];
      }
    }
  }
}

// ---------------------------------------------------------------------------
// Kernel 5: merge chunk softmax partials -> (rowmax, rowsum) per candidate
// ---------------------------------------------------------------------------
__global__ void k_stats2(const float* __restrict__ pmax, const float* __restrict__ psum,
                         float* __restrict__ rstat) {
  const int sb = blockIdx.x;
  const int c = threadIdx.x;
  float gm = -1e30f;
  for (int k = 0; k < 16; ++k) gm = fmaxf(gm, pmax[(sb * 16 + k) * CCAND + c]);
  float gs = 0.0f;
  for (int k = 0; k < 16; ++k)
    gs += psum[(sb * 16 + k) * CCAND + c] * __expf(pmax[(sb * 16 + k) * CCAND + c] - gm);
  rstat[((size_t)sb * CCAND + c) * 2 + 0] = gm;
  rstat[((size_t)sb * CCAND + c) * 2 + 1] = gs;
}

// ---------------------------------------------------------------------------
// Kernel 6: alpha -> blended K/V (WMMA) -> slot EMA updates + strength totals
// ---------------------------------------------------------------------------
__global__ void __launch_bounds__(32) k_write(
    const unsigned short* __restrict__ emKh,
    const float* __restrict__ emK, const float* __restrict__ emV,
    const float* __restrict__ emS, const float* __restrict__ emAge,
    const unsigned short* __restrict__ ckh,
    const unsigned short* __restrict__ ckhT, const unsigned short* __restrict__ cvhT,
    const float* __restrict__ candw, const float* __restrict__ rstat,
    const float* __restrict__ tau, const float* __restrict__ wwp,
    const float* __restrict__ gem, const float* __restrict__ decayp,
    float* __restrict__ newK, float* __restrict__ newV,
    float* __restrict__ newS, float* __restrict__ newA,
    float* __restrict__ totS) {
  const int lane = threadIdx.x;
  const int chunk = blockIdx.x & 15;
  const int sb = blockIdx.x >> 4;
  const int col = lane & 15, hi = lane >> 4, rowoff = hi * 8;
  const float tauc = fmaxf(tau[sb], 0.01f);
  const float wws = wwp[sb];
  const float gg = gem[sb];
  const float dec = decayp[sb];

  __shared__ float candwL[CCAND], gmL[CCAND], gsL[CCAND];
  __shared__ unsigned short aT[16][CCAND + 2];
  __shared__ float bKs[16][DDIM + 1];
  __shared__ float bVs[16][DDIM + 1];

  for (int c = lane; c < CCAND; c += 32) {
    candwL[c] = candw[(size_t)sb * CCAND + c];
    gmL[c] = rstat[((size_t)sb * CCAND + c) * 2 + 0];
    gsL[c] = fmaxf(rstat[((size_t)sb * CCAND + c) * 2 + 1], 1e-30f);
  }

  B16V ak[4][2];
#pragma unroll
  for (int ct = 0; ct < 4; ++ct) {
    const unsigned short* crow = ckh + ((size_t)sb * CCAND + ct * 16 + col) * DDIM;
#pragma unroll
    for (int ks = 0; ks < 2; ++ks) {
      const int d0 = ks * 32 + hi * 16;
#pragma unroll
      for (int j = 0; j < 8; ++j) {
        ak[ct][ks].u[j]     = crow[d0 + j];
        ak[ct][ks].u[j + 8] = crow[d0 + 8 + j];
      }
    }
  }
  __syncthreads();

  const unsigned short* Kb = emKh + (size_t)sb * MMEM * DDIM;
  const float* Sb = emS + (size_t)sb * MMEM;
  float tot = 0.0f;

  for (int t = 0; t < 8; ++t) {
    const int mt = chunk * 8 + t;
    const int m = mt * 16 + col;
    const unsigned short* krow = Kb + (size_t)m * DDIM;
    B16V bk[2];
#pragma unroll
    for (int ks = 0; ks < 2; ++ks) {
      const unsigned short* p = krow + ks * 32 + hi * 16;
      bk[ks].q[0] = *(const uint4*)(p);
      bk[ks].q[1] = *(const uint4*)(p + 8);
    }
    const float Sv = Sb[m];
    const float negS = -Sv * wws;
    float apsv = 0.0f;
#pragma unroll
    for (int ct = 0; ct < 4; ++ct) {
      v8f acc = vzero();
      acc = wmma_bf16(ak[ct][0].v, bk[0].v, acc);
      acc = wmma_bf16(ak[ct][1].v, bk[1].v, acc);
#pragma unroll
      for (int r = 0; r < 8; ++r) {
        const int c = ct * 16 + r + rowoff;
        const float l = (acc[r] + negS) / tauc;
        const float a = gg * candwL[c] * __expf(l - gmL[c]) / gsL[c];
        apsv += a;
        aT[col][c] = f2bf(a);
      }
    }
    apsv += __shfl_xor(apsv, 16, 32);
    __syncthreads();  // aT ready; previous output stage finished

    v8f bK[4], bV[4];
#pragma unroll
    for (int dt = 0; dt < 4; ++dt) { bK[dt] = vzero(); bV[dt] = vzero(); }
#pragma unroll
    for (int ks = 0; ks < 2; ++ks) {
      B16V aa;
      const int k0 = ks * 32 + hi * 16;
#pragma unroll
      for (int j = 0; j < 8; ++j) {
        aa.u[j]     = aT[col][k0 + j];
        aa.u[j + 8] = aT[col][k0 + 8 + j];
      }
#pragma unroll
      for (int dt = 0; dt < 4; ++dt) {
        const size_t base = ((size_t)sb * DDIM + dt * 16 + col) * CCAND + ks * 32 + hi * 16;
        B16V bkc, bvc;
        bkc.q[0] = *(const uint4*)(ckhT + base);
        bkc.q[1] = *(const uint4*)(ckhT + base + 8);
        bvc.q[0] = *(const uint4*)(cvhT + base);
        bvc.q[1] = *(const uint4*)(cvhT + base + 8);
        bK[dt] = wmma_bf16(aa.v, bkc.v, bK[dt]);
        bV[dt] = wmma_bf16(aa.v, bvc.v, bV[dt]);
      }
    }
#pragma unroll
    for (int dt = 0; dt < 4; ++dt) {
#pragma unroll
      for (int r = 0; r < 8; ++r) {
        bKs[r + rowoff][dt * 16 + col] = bK[dt][r];
        bVs[r + rowoff][dt * 16 + col] = bV[dt][r];
      }
    }
    __syncthreads();  // bKs/bVs ready

    const float denom = fmaxf(apsv, EPSV);
    const float ae = fminf(apsv, 1.0f);
    const bool upd = apsv > EPSV;
    const size_t rowbase = ((size_t)sb * MMEM + m) * DDIM;
    if (hi == 0) {
      float nrm = 0.0f;
      for (int d = 0; d < DDIM; ++d) {
        const float v = bKs[col][d] / denom;
        nrm += v * v;
      }
      const float inv = 1.0f / fmaxf(sqrtf(nrm), EPSV);
      for (int d = 0; d < DDIM; ++d) {
        const float bkn = (bKs[col][d] / denom) * inv;
        const float old = emK[rowbase + d];
        newK[rowbase + d] = upd ? (1.0f - ae) * old + ae * bkn : old;
      }
      const float ns = fminf(fmaxf(Sv + apsv, 0.0f), SMAXV) * dec;
      newS[(size_t)sb * MMEM + m] = ns;
      newA[(size_t)sb * MMEM + m] = emAge[(size_t)sb * MMEM + m] * (1.0f - apsv);
      tot += ns;
    } else {
      for (int d = 0; d < DDIM; ++d) {
        const float bvn = bVs[col][d] / denom;
        const float old = emV[rowbase + d];
        newV[rowbase + d] = upd ? (1.0f - ae) * old + ae * bvn : old;
      }
    }
  }
#pragma unroll
  for (int k = 1; k < 32; k <<= 1) tot += __shfl_xor(tot, k, 32);
  if (lane == 0) atomicAdd(&totS[sb], tot);
}

// ---------------------------------------------------------------------------
// Kernel 7: budget enforcement on new_S
// ---------------------------------------------------------------------------
__global__ void k_budget(float* __restrict__ newS, const float* __restrict__ totS) {
  const int i = blockIdx.x * blockDim.x + threadIdx.x;
  if (i < BSZ * NBH * MMEM) {
    const int sb = i / MMEM;
    newS[i] *= fminf(1.0f, BUDGETV / (totS[sb] + EPSV));
  }
}

// ---------------------------------------------------------------------------
extern "C" void kernel_launch(void* const* d_in, const int* in_sizes, int n_in,
                              void* d_out, int out_size, void* d_ws, size_t ws_size,
                              hipStream_t stream) {
  (void)in_sizes; (void)n_in; (void)out_size; (void)ws_size;
  const float* q        = (const float*)d_in[0];
  const float* q_nov    = (const float*)d_in[1];
  const float* v_nov    = (const float*)d_in[2];
  const float* surprise = (const float*)d_in[3];
  const float* w_nov    = (const float*)d_in[4];
  const float* g_em     = (const float*)d_in[5];
  const float* tau      = (const float*)d_in[6];
  const float* decay    = (const float*)d_in[7];
  const float* ww       = (const float*)d_in[8];
  const float* emK      = (const float*)d_in[9];
  const float* emV      = (const float*)d_in[10];
  const float* emS      = (const float*)d_in[11];
  const float* emAge    = (const float*)d_in[12];

  float* outp = (float*)d_out;
  float* newK = outp + (size_t)BSZ * NSEQ * NBH * DDIM;
  float* newV = newK + (size_t)BSZ * NBH * MMEM * DDIM;
  float* newS = newV + (size_t)BSZ * NBH * MMEM * DDIM;
  float* newA = newS + (size_t)BSZ * NBH * MMEM;

  char* wp = (char*)d_ws;
  unsigned short* emKh  = (unsigned short*)wp; wp += (size_t)BSZ * NBH * MMEM * DDIM * 2;
  unsigned short* emVhT = (unsigned short*)wp; wp += (size_t)BSZ * NBH * MMEM * DDIM * 2;
  unsigned short* ckh   = (unsigned short*)wp; wp += (size_t)BSZ * NBH * CCAND * DDIM * 2;
  unsigned short* ckhT  = (unsigned short*)wp; wp += (size_t)BSZ * NBH * CCAND * DDIM * 2;
  unsigned short* cvhT  = (unsigned short*)wp; wp += (size_t)BSZ * NBH * CCAND * DDIM * 2;
  float* nov   = (float*)wp; wp += (size_t)BSZ * NBH * NSEQ * 4;
  float* candw = (float*)wp; wp += (size_t)BSZ * NBH * CCAND * 4;
  float* pmax  = (float*)wp; wp += (size_t)BSZ * NBH * 16 * CCAND * 4;
  float* psum  = (float*)wp; wp += (size_t)BSZ * NBH * 16 * CCAND * 4;
  float* rstat = (float*)wp; wp += (size_t)BSZ * NBH * CCAND * 2 * 4;
  float* totS  = (float*)wp; wp += (size_t)BSZ * NBH * 4;

  k_convert<<<1024, 256, 0, stream>>>(emK, emV, emKh, emVhT);
  k_attn<<<BSZ * NBH * (NSEQ / 16), 32, 0, stream>>>(q, q_nov, surprise, w_nov, emS,
                                                     emKh, emVhT, outp, nov);
  k_cand<<<BSZ * NBH, 32, 0, stream>>>(nov, q_nov, v_nov, ckh, ckhT, cvhT, candw, totS);
  k_stats1<<<BSZ * NBH * 16, 32, 0, stream>>>(emKh, emS, ckh, tau, ww, pmax, psum);
  k_stats2<<<BSZ * NBH, CCAND, 0, stream>>>(pmax, psum, rstat);
  k_write<<<BSZ * NBH * 16, 32, 0, stream>>>(emKh, emK, emV, emS, emAge, ckh, ckhT, cvhT,
                                             candw, rstat, tau, ww, g_em, decay,
                                             newK, newV, newS, newA, totS);
  k_budget<<<(BSZ * NBH * MMEM + 255) / 256, 256, 0, stream>>>(newS, totS);
}
